// MultiHeadAttention_24223615550206
// MI455X (gfx1250) — matrix operand order repaired and checked by reading, not compiled, not run on hardware
//
#include <hip/hip_runtime.h>
#include <hip/hip_bf16.h>

// ---------------------------------------------------------------------------
// MHA forward for MI455X (gfx1250): bf16 WMMA everywhere, flash-style softmax.
//   B=2, L=2048, D=1024, H=16, DK=64.
// Outputs (concatenated): out [B,L,D] f32, k [B,H,L,DK] f32, v [B,H,L,DK] f32.
// ---------------------------------------------------------------------------

#define B_  2
#define L_  2048
#define D_  1024
#define H_  16
#define DK_ 64

typedef __attribute__((ext_vector_type(16))) __bf16 v16bf;
typedef __attribute__((ext_vector_type(8)))  float  v8f;

union Frag {
    v16bf v;
    uint4 q[2];
};

__device__ __forceinline__ uint32_t bf_round(float x) {
    uint32_t u = __builtin_bit_cast(uint32_t, x);
    return (u + 0x7FFFu + ((u >> 16) & 1u)) >> 16;   // round-to-nearest-even
}
__device__ __forceinline__ uint32_t pack_bf2(float lo, float hi) {
    return bf_round(lo) | (bf_round(hi) << 16);
}
__device__ __forceinline__ uint16_t bf1(float x) { return (uint16_t)bf_round(x); }

__device__ __forceinline__ v8f wmma_bf16(const Frag& a, const Frag& b, v8f c) {
    return __builtin_amdgcn_wmma_f32_16x16x32_bf16(false, a.v, false, b.v,
                                                   (short)0, c, false, false);
}

// ---------------------------------------------------------------------------
// GEMM: C[M x N] = A[M x 1024] * W[1024 x N] + bias
// MODE 0: A = x (f32), N = 3072, epilogue routes q/k/v.
// MODE 1: A = ctx (bf16), N = 1024, epilogue writes f32 out.
// Block: 256 threads (8 waves), tile 64(M) x 128(N), BK = 32.
// ---------------------------------------------------------------------------
template <int MODE>
__global__ __launch_bounds__(256) void gemm_kernel(
    const void* __restrict__ Aip, const float* __restrict__ W,
    const float* __restrict__ bias,
    float* __restrict__ outF,                       // MODE 1
    float* __restrict__ kcache, float* __restrict__ vcache,  // MODE 0
    uint16_t* __restrict__ qbf, uint16_t* __restrict__ kbf,
    uint16_t* __restrict__ vtbf,                    // MODE 0 (V transposed)
    int N)
{
    __shared__ uint32_t lA[64 * 16];    // 64 rows x 16 dwords (32 bf16 K)
    __shared__ uint32_t lB[128 * 16];   // 128 cols x 16 dwords (32 bf16 K)

    const int nb   = blockIdx.x * 128;
    const int mb   = blockIdx.y * 64;
    const int tid  = threadIdx.x;
    const int wave = tid >> 5;
    const int lane = tid & 31;
    const int half = lane >> 4;
    const int ln   = lane & 15;
    const int wm   = (wave >> 2) * 32;  // wave row offset in tile
    const int wn   = (wave & 3) * 32;   // wave col offset in tile

    v8f acc[2][2] = {};

    for (int k0 = 0; k0 < 1024; k0 += 32) {
        // --- stage A chunk (64 x 32) as bf16, row-major dwords ---
        for (int i = 0; i < 4; ++i) {
            int slot = tid + i * 256;            // 0..1023
            int m = slot >> 4, kd = slot & 15;
            uint32_t val;
            if (MODE == 0) {
                const float* ap = (const float*)Aip + (size_t)(mb + m) * 1024 + k0 + kd * 2;
                float2 f = *(const float2*)ap;
                val = pack_bf2(f.x, f.y);
            } else {
                const uint32_t* ap = (const uint32_t*)Aip + (size_t)(mb + m) * 512 + (k0 >> 1) + kd;
                val = *ap;
            }
            lA[slot] = val;
        }
        // --- stage B chunk (32 x 128): lB[n*16+kd] = pack(W[k0+2kd][n], W[k0+2kd+1][n]) ---
        for (int i = 0; i < 8; ++i) {
            int slot = tid + i * 256;            // 0..2047
            int n = slot & 127, kd = slot >> 7;
            const float* wp = W + (size_t)(k0 + 2 * kd) * N + nb + n;
            lB[n * 16 + kd] = pack_bf2(wp[0], wp[N]);
        }
        __syncthreads();

        // --- fragments (b128 loads in WMMA layout) ---
        Frag a[2], b[2];
        for (int r = 0; r < 2; ++r) {
            const uint32_t* p = &lA[(wm + r * 16 + ln) * 16];
            a[r].q[0] = *(const uint4*)(p + half * 4);
            a[r].q[1] = *(const uint4*)(p + half * 4 + 8);
        }
        for (int c = 0; c < 2; ++c) {
            const uint32_t* p = &lB[(wn + c * 16 + ln) * 16];
            b[c].q[0] = *(const uint4*)(p + half * 4);
            b[c].q[1] = *(const uint4*)(p + half * 4 + 8);
        }
        for (int r = 0; r < 2; ++r)
            for (int c = 0; c < 2; ++c)
                acc[r][c] = wmma_bf16(a[r], b[c], acc[r][c]);
        __syncthreads();
    }

    // --- epilogue ---
    for (int r = 0; r < 2; ++r)
        for (int c = 0; c < 2; ++c) {
            int col = nb + wn + c * 16 + ln;
            float bv = bias[col];
            for (int vr = 0; vr < 8; ++vr) {
                int row = mb + wm + r * 16 + vr + half * 8;  // 0..4095 = b*L + l
                float val = acc[r][c][vr] + bv;
                if (MODE == 1) {
                    outF[(size_t)row * 1024 + col] = val;
                } else {
                    int bi = row >> 11, l = row & 2047;
                    int which = col >> 10, cc = col & 1023;
                    int h = cc >> 6, dk = cc & 63;
                    size_t hoff = ((size_t)(bi * H_ + h) * L_ + l) * DK_ + dk;
                    uint16_t bvv = bf1(val);
                    if (which == 0) {
                        qbf[hoff] = bvv;
                    } else if (which == 1) {
                        kcache[hoff] = val;
                        kbf[hoff] = bvv;
                    } else {
                        vcache[hoff] = val;
                        vtbf[((size_t)(bi * H_ + h) * DK_ + dk) * L_ + l] = bvv;
                    }
                }
            }
        }
}

// ---------------------------------------------------------------------------
// Flash attention: 1 wave per (b, h, 16-query tile). Online softmax over
// 32-key blocks; causal via loop bound + ragged-block mask.
// ---------------------------------------------------------------------------
__global__ __launch_bounds__(32) void attn_kernel(
    const uint32_t* __restrict__ qbf,   // [B,H,L,DK] bf16 as dwords
    const uint32_t* __restrict__ kbf,   // [B,H,L,DK] bf16 as dwords
    const uint32_t* __restrict__ vtbf,  // [B,H,DK,L] bf16 as dwords
    uint16_t* __restrict__ ctxbf)       // [B,L,D] bf16
{
    __shared__ uint32_t lP[16 * 16];    // P tile: 16 rows x 32 keys (bf16)

    const int bid   = blockIdx.x;
    const int qt    = bid & 127;        // L/16 tiles
    const int bh    = bid >> 7;         // 0..31
    const int qbase = qt * 16;
    const int lane  = threadIdx.x;
    const int half  = lane >> 4;
    const int ln    = lane & 15;

    // Q fragments (persist in registers): row = 32 dwords of bf16 DK values
    const uint32_t* qrow = qbf + ((size_t)bh * L_ + qbase + ln) * 32;
    Frag qf[2];
    for (int ks = 0; ks < 2; ++ks) {
        qf[ks].q[0] = *(const uint4*)(qrow + ks * 16 + half * 4);
        qf[ks].q[1] = *(const uint4*)(qrow + ks * 16 + half * 4 + 8);
    }

    v8f acc[4] = {};
    float mrun[8], srun[8];
    for (int r = 0; r < 8; ++r) { mrun[r] = -1e30f; srun[r] = 0.0f; }

    const int nkb = (qbase + 16 + 31) >> 5;   // causal key-block count
    for (int kb = 0; kb < nkb * 32; kb += 32) {
        // K^T B-fragments: lane n = key, VGPR slots = dk pairs
        Frag kf[2][2];
        for (int nt = 0; nt < 2; ++nt) {
            const uint32_t* kr = kbf + ((size_t)bh * L_ + kb + nt * 16 + ln) * 32;
            for (int ks = 0; ks < 2; ++ks) {
                kf[nt][ks].q[0] = *(const uint4*)(kr + ks * 16 + half * 4);
                kf[nt][ks].q[1] = *(const uint4*)(kr + ks * 16 + half * 4 + 8);
            }
        }
        // S = Q K^T  (accumulate over DK in 2 WMMA k-steps per 16-key tile)
        v8f s0 = {}, s1 = {};
        s0 = wmma_bf16(qf[0], kf[0][0], s0);
        s0 = wmma_bf16(qf[1], kf[0][1], s0);
        s1 = wmma_bf16(qf[0], kf[1][0], s1);
        s1 = wmma_bf16(qf[1], kf[1][1], s1);

        const float scale = 0.125f;     // 1/sqrt(64)
        const bool maskblk = (kb + 31 > qbase);
        for (int vr = 0; vr < 8; ++vr) {
            float a0 = s0[vr] * scale, a1 = s1[vr] * scale;
            if (maskblk) {
                int qr = qbase + vr + half * 8;
                if (kb + ln > qr)      a0 = -1e30f;
                if (kb + 16 + ln > qr) a1 = -1e30f;
            }
            s0[vr] = a0; s1[vr] = a1;
        }

        // Online softmax per row (rows split by half-wave; 16-lane butterflies)
        float p0[8], p1[8];
        for (int vr = 0; vr < 8; ++vr) {
            float t = fmaxf(s0[vr], s1[vr]);
            for (int off = 1; off < 16; off <<= 1)
                t = fmaxf(t, __shfl_xor(t, off, 32));
            float mnew = fmaxf(mrun[vr], t);
            float sc = __expf(mrun[vr] - mnew);
            float e0 = __expf(s0[vr] - mnew);
            float e1 = __expf(s1[vr] - mnew);
            float rs = e0 + e1;
            for (int off = 1; off < 16; off <<= 1)
                rs += __shfl_xor(rs, off, 32);
            srun[vr] = srun[vr] * sc + rs;
            mrun[vr] = mnew;
            for (int t2 = 0; t2 < 4; ++t2) acc[t2][vr] *= sc;
            p0[vr] = e0; p1[vr] = e1;
        }

        // P (C layout) -> LDS row-major bf16 -> reload as A fragment
        uint16_t* lp16 = (uint16_t*)lP;
        for (int vr = 0; vr < 8; ++vr) {
            int row = vr + half * 8;
            lp16[row * 32 + ln]      = bf1(p0[vr]);
            lp16[row * 32 + 16 + ln] = bf1(p1[vr]);
        }
        __syncthreads();
        Frag pf;
        {
            const uint32_t* pr = lP + ln * 16;
            pf.q[0] = *(const uint4*)(pr + half * 4);
            pf.q[1] = *(const uint4*)(pr + half * 4 + 8);
        }
        // ctx += P * V  (V transposed: key pairs contiguous)
        for (int t = 0; t < 4; ++t) {
            Frag vf;
            const uint32_t* vr_ = vtbf + ((size_t)bh * DK_ + t * 16 + ln) * (L_ / 2) + (kb >> 1);
            vf.q[0] = *(const uint4*)(vr_ + half * 4);
            vf.q[1] = *(const uint4*)(vr_ + half * 4 + 8);
            acc[t] = wmma_bf16(pf, vf, acc[t]);
        }
        __syncthreads();
    }

    // ctx / rowsum -> bf16 ctx buffer [B, L, D] at col h*64 + t*16 + ln
    const int bi = bh >> 4, h = bh & 15;
    for (int t = 0; t < 4; ++t)
        for (int vr = 0; vr < 8; ++vr) {
            int row = vr + half * 8;
            float v = acc[t][vr] / srun[vr];
            size_t off = ((size_t)bi * L_ + qbase + row) * D_ + h * DK_ + t * 16 + ln;
            ctxbf[off] = bf1(v);
        }
}

// ---------------------------------------------------------------------------
extern "C" void kernel_launch(void* const* d_in, const int* in_sizes, int n_in,
                              void* d_out, int out_size, void* d_ws, size_t ws_size,
                              hipStream_t stream) {
    (void)in_sizes; (void)n_in; (void)out_size; (void)ws_size;
    const float* x     = (const float*)d_in[0];
    const float* w_qkv = (const float*)d_in[1];
    const float* b_qkv = (const float*)d_in[2];
    const float* w_o   = (const float*)d_in[3];
    const float* b_o   = (const float*)d_in[4];
    // d_in[5] = mask: causal handled analytically.

    float* out    = (float*)d_out;
    float* kcache = out + (size_t)B_ * L_ * D_;
    float* vcache = kcache + (size_t)B_ * L_ * D_;

    uint8_t* ws = (uint8_t*)d_ws;
    const size_t HBYTES = (size_t)B_ * H_ * L_ * DK_ * 2;   // 8 MB each
    uint16_t* qbf   = (uint16_t*)(ws);
    uint16_t* kbf   = (uint16_t*)(ws + HBYTES);
    uint16_t* vtbf  = (uint16_t*)(ws + 2 * HBYTES);
    uint16_t* ctxbf = (uint16_t*)(ws + 3 * HBYTES);

    dim3 blk(256);
    // 1) QKV projection: M=4096, N=3072
    gemm_kernel<0><<<dim3(3072 / 128, 4096 / 64), blk, 0, stream>>>(
        x, w_qkv, b_qkv, nullptr, kcache, vcache, qbf, kbf, vtbf, 3072);
    // 2) Flash attention: one wave per (b, h, 16-query tile)
    attn_kernel<<<dim3(B_ * H_ * (L_ / 16)), dim3(32), 0, stream>>>(
        (const uint32_t*)qbf, (const uint32_t*)kbf, (const uint32_t*)vtbf, ctxbf);
    // 3) Output projection: M=4096, N=1024
    gemm_kernel<1><<<dim3(1024 / 128, 4096 / 64), blk, 0, stream>>>(
        ctxbf, w_o, b_o, out, nullptr, nullptr, nullptr, nullptr, nullptr, 1024);
}
